// Dynamic_Offset_Extimator_15444702396602
// MI455X (gfx1250) — matrix operand order repaired and checked by reading, not compiled, not run on hardware
//
#include <hip/hip_runtime.h>
#include <hip/hip_bf16.h>

// Non-local block (self-attention): B=8, C=64, N=4096, Ci=32.
// Flash-attention fusion: never materialize the 512MB f matrix.
// bf16 WMMA (16x16x32) with fp32 accumulate; K=32 == Ci exactly.
//  - theta pre-scaled by log2(e) -> all softmax exps are bare v_exp_f32
//  - softmax row sums via WMMA against all-ones B fragment
//  - row max via DPP16 butterfly (pure VALU)
//  - double-buffered phi/g fragments (loads overlap a full iteration)
//  - P stored to LDS as packed bf16x2 in a permuted key order; gT is stored
//    pre-permuted at projection time so g fragment loads are contiguous

typedef __bf16 bf16x2  __attribute__((ext_vector_type(2)));
typedef __bf16 bf16x8  __attribute__((ext_vector_type(8)));
typedef __bf16 bf16x16 __attribute__((ext_vector_type(16)));
typedef float  f32x8   __attribute__((ext_vector_type(8)));
typedef float  f32x4   __attribute__((ext_vector_type(4)));

#define NSP 4096   // spatial positions H*W
#define CI  32     // inter channels
#define CC  64     // in channels
#define NB  8      // batch
#define LOG2E 1.44269504088896340736f

static __device__ __forceinline__ bf16x16 cat8(bf16x8 lo, bf16x8 hi) {
  return __builtin_shufflevector(lo, hi, 0,1,2,3,4,5,6,7,8,9,10,11,12,13,14,15);
}

#define WMMA_BF16(A, B, C) \
  __builtin_amdgcn_wmma_f32_16x16x32_bf16(false, (A), false, (B), (short)0, (C), false, false)

// --- DPP16 butterfly max over each 16-lane row (lanes 0-15 and 16-31). ---
template <int CTRL>
static __device__ __forceinline__ float dppmax(float v) {
  const int t = __builtin_amdgcn_update_dpp(0, __float_as_int(v), CTRL, 0xF, 0xF, true);
  return fmaxf(v, __int_as_float(t));
}
static __device__ __forceinline__ float rowmax16(float v) {
  v = dppmax<0xB1>(v);   // quad_perm(1,0,3,2)  : xor 1
  v = dppmax<0x4E>(v);   // quad_perm(2,3,0,1)  : xor 2
  v = dppmax<0x141>(v);  // row_half_mirror     : combine quads
  v = dppmax<0x140>(v);  // row_mirror          : combine octets
  return v;
}

// ---------------------------------------------------------------------------
// Kernel 1: fused 1x1-conv projections.
//   theta (N x 32) bf16 row-major, pre-scaled by log2(e)
//   phiT  (N x 32) bf16 row-major
//   gT    (32 x N) bf16 row-major with each 32-key block stored in the
//          interleaved order (key c -> slot 2c, key 16+c -> slot 2c+1) that
//          matches the packed P write in the flash kernel.
// ---------------------------------------------------------------------------
__global__ __launch_bounds__(256)
void nl_proj_kernel(const float* __restrict__ x,
                    const float* __restrict__ Wg, const float* __restrict__ bg,
                    const float* __restrict__ Wt, const float* __restrict__ bt,
                    const float* __restrict__ Wp, const float* __restrict__ bp,
                    __bf16* __restrict__ theta, __bf16* __restrict__ phiT,
                    __bf16* __restrict__ gT) {
  __shared__ float xs[64][65];               // 64 channels x 64 positions (+pad)
  const int b   = blockIdx.x >> 6;           // 8 batches
  const int n0  = (blockIdx.x & 63) * 64;    // 64-wide spatial tile
  const int tid = threadIdx.x;

  for (int i = 0; i < 16; ++i) {
    int lin = tid + i * 256;
    int c = lin >> 6, nl = lin & 63;
    xs[c][nl] = x[((size_t)b * CC + c) * NSP + n0 + nl];
  }
  __syncthreads();

  const int nl = tid & 63;
  const int dg = tid >> 6;                   // 4 groups of 8 output channels
  // permuted slot within the 32-key block for the g layout
  const int kc = nl & 31;
  const int kp = (kc < 16) ? (2 * kc) : (2 * (kc - 16) + 1);
  const int gcol = (nl & 32) + kp;           // position within the 64-wide tile

  for (int dd = 0; dd < 8; ++dd) {
    const int d = dg * 8 + dd;               // 0..31
    float at = bt[d], ap = bp[d], ag = bg[d];
#pragma unroll 8
    for (int c = 0; c < 64; ++c) {
      const float xv = xs[c][nl];
      at = __builtin_fmaf(Wt[d * 64 + c], xv, at);
      ap = __builtin_fmaf(Wp[d * 64 + c], xv, ap);
      ag = __builtin_fmaf(Wg[d * 64 + c], xv, ag);
    }
    const size_t nIdx = (size_t)b * NSP + n0 + nl;
    theta[nIdx * CI + d] = (__bf16)(at * LOG2E);   // log2-domain scores
    phiT [nIdx * CI + d] = (__bf16)ap;
    gT[((size_t)b * CI + d) * NSP + n0 + gcol] = (__bf16)ag;
  }
}

// ---------------------------------------------------------------------------
// Kernel 2: flash attention. One wave32 per 32-query block; 32-key steps.
// ---------------------------------------------------------------------------
__global__ __launch_bounds__(64)
void nl_flash_kernel(const __bf16* __restrict__ theta,
                     const __bf16* __restrict__ phiT,
                     const __bf16* __restrict__ gT,
                     float* __restrict__ y) {
  __shared__ __bf16 Pt[2][32 * 32];          // per-wave P bounce buffer (2KB each)
  const int wave   = threadIdx.x >> 5;
  const int lane   = threadIdx.x & 31;
  const int laneHi = lane >> 4;
  const int laneLo = lane & 15;

  const int rb = blockIdx.x * 2 + wave;      // 1024 row blocks total
  const int b  = rb >> 7;                    // 128 blocks per batch
  const int q0 = (rb & 127) * 32;

  const __bf16* thB = theta + (size_t)b * NSP * CI;
  const __bf16* phB = phiT  + (size_t)b * NSP * CI;
  const __bf16* gB  = gT    + (size_t)b * CI  * NSP;

  // --- A tiles (queries), loop invariant.
  bf16x16 Aq0, Aq1;
  {
    const __bf16* r0 = thB + (size_t)(q0 + laneLo) * CI;
    const __bf16* r1 = thB + (size_t)(q0 + 16 + laneLo) * CI;
    Aq0 = cat8(*(const bf16x8*)(r0 + laneHi * 8),
               *(const bf16x8*)(r0 + 16 + laneHi * 8));
    Aq1 = cat8(*(const bf16x8*)(r1 + laneHi * 8),
               *(const bf16x8*)(r1 + 16 + laneHi * 8));
  }

  // All-ones B fragment for the row-sum WMMA; pin in VGPRs so it is not
  // rematerialized from SGPRs every iteration (WMMA sources must be VGPRs).
  bf16x16 Bones;
#pragma unroll
  for (int i = 0; i < 16; ++i) Bones[i] = (__bf16)1.0f;
  asm volatile("" : "+v"(Bones));

  float m0[8], m1[8];
  f32x8 O00 = {}, O01 = {}, O10 = {}, O11 = {};
  f32x8 L0  = {}, L1  = {};
#pragma unroll
  for (int j = 0; j < 8; ++j) {
    m0[j] = -__builtin_inff(); m1[j] = -__builtin_inff();
  }

  __bf16* myP = &Pt[wave][0];

  // phi B tile: lane col c = laneLo (+16); elems 0..7 -> K = laneHi*8 + h, 8..15 -> K = 16 + laneHi*8 + (h-8).
  auto loadPhi = [&](int k, bf16x16& T0, bf16x16& T1) {
    const __bf16* pc0 = phB + (size_t)(k + laneLo) * CI + laneHi * 8;
    const __bf16* pc1 = phB + (size_t)(k + 16 + laneLo) * CI + laneHi * 8;
    T0 = cat8(*(const bf16x8*)pc0, *(const bf16x8*)(pc0 + 16));
    T1 = cat8(*(const bf16x8*)pc1, *(const bf16x8*)(pc1 + 16));
  };
  // g B tile: gT is pre-permuted, so this is a plain contiguous load.
  auto loadG = [&](int k, bf16x16& T0, bf16x16& T1) {
    const __bf16* g0p = gB + (size_t)laneLo * NSP + k + laneHi * 16;
    const __bf16* g1p = gB + (size_t)(16 + laneLo) * NSP + k + laneHi * 16;
    T0 = cat8(*(const bf16x8*)g0p, *(const bf16x8*)(g0p + 8));
    T1 = cat8(*(const bf16x8*)g1p, *(const bf16x8*)(g1p + 8));
  };

  // Software pipeline: prime the first key block.
  bf16x16 B0, B1, G0, G1;
  loadPhi(0, B0, B1);
  loadG(0, G0, G1);

  for (int kb = 0; kb < NSP; kb += 32) {
    // Issue next block's loads first: a full iteration of WMMA+VALU covers them.
    const int kn = (kb + 32) & (NSP - 1);
    bf16x16 nB0, nB1, nG0, nG1;
    loadPhi(kn, nB0, nB1);
    loadG(kn, nG0, nG1);
    __builtin_prefetch(phB + (size_t)(kn + 32 + laneLo) * CI, 0, 1);
    __builtin_prefetch(gB + (size_t)laneLo * NSP + kn + 32, 0, 1);

    const f32x8 z = {};
    f32x8 S00 = WMMA_BF16(Aq0, B0, z);
    f32x8 S01 = WMMA_BF16(Aq0, B1, z);
    f32x8 S10 = WMMA_BF16(Aq1, B0, z);
    f32x8 S11 = WMMA_BF16(Aq1, B1, z);

    // --- online softmax (log2 domain), q-rows 0..15.
#pragma unroll
    for (int j = 0; j < 8; ++j) {
      const float v  = rowmax16(fmaxf(S00[j], S01[j]));
      const float mn = fmaxf(m0[j], v);
      const float a  = __builtin_amdgcn_exp2f(m0[j] - mn);
      m0[j] = mn;
      const float p0 = __builtin_amdgcn_exp2f(S00[j] - mn);
      const float p1 = __builtin_amdgcn_exp2f(S01[j] - mn);
      O00[j] *= a; O01[j] *= a; L0[j] *= a;
      bf16x2 pk; pk.x = (__bf16)p0; pk.y = (__bf16)p1;
      *(bf16x2*)(myP + (laneHi * 8 + j) * 32 + laneLo * 2) = pk;
    }
    // --- q-rows 16..31.
#pragma unroll
    for (int j = 0; j < 8; ++j) {
      const float v  = rowmax16(fmaxf(S10[j], S11[j]));
      const float mn = fmaxf(m1[j], v);
      const float a  = __builtin_amdgcn_exp2f(m1[j] - mn);
      m1[j] = mn;
      const float p0 = __builtin_amdgcn_exp2f(S10[j] - mn);
      const float p1 = __builtin_amdgcn_exp2f(S11[j] - mn);
      O10[j] *= a; O11[j] *= a; L1[j] *= a;
      bf16x2 pk; pk.x = (__bf16)p0; pk.y = (__bf16)p1;
      *(bf16x2*)(myP + (16 + laneHi * 8 + j) * 32 + laneLo * 2) = pk;
    }

    // --- re-read P in A layout from LDS (wave-private; DScnt ordering).
    const __bf16* pr0 = myP + laneLo * 32;
    const __bf16* pr1 = myP + (16 + laneLo) * 32;
    const bf16x16 AP0 = cat8(*(const bf16x8*)(pr0 + laneHi * 16),
                             *(const bf16x8*)(pr0 + laneHi * 16 + 8));
    const bf16x16 AP1 = cat8(*(const bf16x8*)(pr1 + laneHi * 16),
                             *(const bf16x8*)(pr1 + laneHi * 16 + 8));

    // denominators: L += P * ones
    L0 = WMMA_BF16(AP0, Bones, L0);
    L1 = WMMA_BF16(AP1, Bones, L1);

    O00 = WMMA_BF16(AP0, G0, O00);
    O01 = WMMA_BF16(AP0, G1, O01);
    O10 = WMMA_BF16(AP1, G0, O10);
    O11 = WMMA_BF16(AP1, G1, O11);

    B0 = nB0; B1 = nB1; G0 = nG0; G1 = nG1;
  }

  // --- normalize and write y (N x 32, fp32).
#pragma unroll
  for (int j = 0; j < 8; ++j) {
    const float inv0 = 1.0f / L0[j];
    const float inv1 = 1.0f / L1[j];
    const int r0 = q0 + laneHi * 8 + j;
    const int r1 = q0 + 16 + laneHi * 8 + j;
    float* y0 = y + ((size_t)b * NSP + r0) * CI;
    float* y1 = y + ((size_t)b * NSP + r1) * CI;
    y0[laneLo]      = O00[j] * inv0;
    y0[16 + laneLo] = O01[j] * inv0;
    y1[laneLo]      = O10[j] * inv1;
    y1[16 + laneLo] = O11[j] * inv1;
  }
}

// ---------------------------------------------------------------------------
// Kernel 3: W_y projection + residual.
// ---------------------------------------------------------------------------
__global__ __launch_bounds__(256)
void nl_out_kernel(const float* __restrict__ x, const float* __restrict__ Ww,
                   const float* __restrict__ bw, const float* __restrict__ y,
                   float* __restrict__ out) {
  const int b = blockIdx.x >> 4;
  const int n = (blockIdx.x & 15) * 256 + threadIdx.x;
  const float* yr = y + ((size_t)b * NSP + n) * CI;
  float yv[CI];
#pragma unroll
  for (int d = 0; d < CI; d += 4) {
    const f32x4 t = *(const f32x4*)(yr + d);
    yv[d] = t[0]; yv[d + 1] = t[1]; yv[d + 2] = t[2]; yv[d + 3] = t[3];
  }
  for (int o = 0; o < CC; ++o) {
    float acc = bw[o];
#pragma unroll 8
    for (int d = 0; d < CI; ++d)
      acc = __builtin_fmaf(Ww[o * CI + d], yv[d], acc);
    const size_t idx = ((size_t)b * CC + o) * NSP + n;
    out[idx] = acc + x[idx];
  }
}

// ---------------------------------------------------------------------------
extern "C" void kernel_launch(void* const* d_in, const int* in_sizes, int n_in,
                              void* d_out, int out_size, void* d_ws, size_t ws_size,
                              hipStream_t stream) {
  const float* x  = (const float*)d_in[0];
  const float* Wg = (const float*)d_in[1];
  const float* bg = (const float*)d_in[2];
  const float* Wt = (const float*)d_in[3];
  const float* bt = (const float*)d_in[4];
  const float* Wp = (const float*)d_in[5];
  const float* bp = (const float*)d_in[6];
  const float* Ww = (const float*)d_in[7];
  const float* bw = (const float*)d_in[8];
  float* out = (float*)d_out;

  const size_t projBytes = (size_t)NB * NSP * CI * sizeof(__bf16);   // 2 MB each
  char* ws = (char*)d_ws;
  __bf16* theta = (__bf16*)(ws);
  __bf16* phiT  = (__bf16*)(ws + projBytes);
  __bf16* gT    = (__bf16*)(ws + 2 * projBytes);
  float*  y     = (float*)(ws + 3 * projBytes);                      // 4 MB

  (void)in_sizes; (void)n_in; (void)out_size; (void)ws_size;

  nl_proj_kernel<<<NB * (NSP / 64), 256, 0, stream>>>(
      x, Wg, bg, Wt, bt, Wp, bp, theta, phiT, gT);

  nl_flash_kernel<<<(NB * (NSP / 32)) / 2, 64, 0, stream>>>(theta, phiT, gT, y);

  nl_out_kernel<<<NB * (NSP / 256), 256, 0, stream>>>(x, Ww, bw, y, out);
}
